// Attention_84756884619871
// MI455X (gfx1250) — matrix operand order repaired and checked by reading, not compiled, not run on hardware
//
#include <hip/hip_runtime.h>
#include <hip/hip_bf16.h>

typedef __attribute__((ext_vector_type(16))) __bf16         v16bf;
typedef __attribute__((ext_vector_type(8)))  float          v8f;
typedef __attribute__((ext_vector_type(16))) unsigned short v16u;
typedef __attribute__((ext_vector_type(8)))  unsigned short v8u;
typedef __attribute__((ext_vector_type(4)))  unsigned int   u32x4;
typedef __attribute__((ext_vector_type(8)))  int            i32x8;
typedef __attribute__((ext_vector_type(4)))  int            i32x4;

#define D_MODEL  2048
#define N_HEADS  16
#define HEAD_DIM 128
#define SEQ      2048
#define BATCH    2
#define ROWS     (BATCH * SEQ)     /* 4096 */
#define QKV_N    (3 * D_MODEL)     /* 6144 */

__device__ __forceinline__ unsigned short f2bf(float f) {
    unsigned u = __builtin_bit_cast(unsigned, f);
    unsigned r = u + 0x7FFFu + ((u >> 16) & 1u);   // round-to-nearest-even
    return (unsigned short)(r >> 16);
}
__device__ __forceinline__ float bf2f(unsigned short h) {
    unsigned u = ((unsigned)h) << 16;
    return __builtin_bit_cast(float, u);
}
__device__ __forceinline__ v16u cat8(v8u lo, v8u hi) {
    return __builtin_shufflevector(lo, hi, 0,1,2,3,4,5,6,7,8,9,10,11,12,13,14,15);
}
__device__ __forceinline__ v16bf as_bf(v16u v) {
    return __builtin_bit_cast(v16bf, v);
}

// ---------------------------------------------------------------------------
// Tensor Data Mover: 2-D bf16 tile (global, row-stride) -> LDS (packed).
// D# per CDNA5 ISA 8.3/8.4. Tensor dims == tile dims (no OOB in this app),
// stride0 = true row pitch in elements. Groups 2/3 zero (<=2-D tensor).
// This toolchain exposes the 6-arg builtin (extra int32x8 descriptor group).
// Issued by one wave; completion via s_wait_tensorcnt.
// ---------------------------------------------------------------------------
__device__ __forceinline__ void tdm_load_2d(unsigned lds_addr, const void* gptr,
                                            unsigned tile_d0, unsigned tile_d1,
                                            unsigned long long stride0)
{
    unsigned long long ga = (unsigned long long)(size_t)gptr;
    u32x4 g0;
    g0.x = 1u;                                                // count=1
    g0.y = lds_addr;                                          // LDS byte addr
    g0.z = (unsigned)(ga & 0xFFFFFFFFu);                      // global addr lo
    g0.w = (unsigned)((ga >> 32) & 0x1FFFFFFu) | (2u << 30);  // addr hi | type=2
    i32x8 g1;
    g1[0] = (int)(1u << 16);                                  // data_size = 2B
    g1[1] = (int)((tile_d0 & 0xFFFFu) << 16);                 // tensor_dim0 lo16
    g1[2] = (int)((tile_d0 >> 16) | ((tile_d1 & 0xFFFFu) << 16)); // td0 hi | td1 lo
    g1[3] = (int)((tile_d1 >> 16) | (tile_d0 << 16));         // td1 hi | tile_dim0
    g1[4] = (int)(tile_d1 & 0xFFFFu);                         // tile_dim1 (dim2=0)
    g1[5] = (int)(stride0 & 0xFFFFFFFFu);                     // stride0 lo32
    g1[6] = (int)((stride0 >> 32) & 0xFFFFu);                 // stride0 hi16
    g1[7] = 0;
    i32x4 z4 = {0, 0, 0, 0};
    i32x8 z8 = {0, 0, 0, 0, 0, 0, 0, 0};
    __builtin_amdgcn_tensor_load_to_lds(g0, g1, z4, z4, z8, 0);
}
__device__ __forceinline__ unsigned lds_off(const void* p) {
    return (unsigned)(size_t)p;    // LDS aperture: addr[31:0] = LDS offset
}

// ---------------------------------------------------------------------------
// One-time converters: fp32 -> bf16 (x), and fp32 [K][N] -> bf16 [N][K]
// (weights transposed so GEMM B-staging is a contiguous row copy for the TDM).
// ---------------------------------------------------------------------------
__global__ __launch_bounds__(256)
void cvt_bf16(const float* __restrict__ in, unsigned short* __restrict__ out, int n)
{
    int i = blockIdx.x * 256 + threadIdx.x;
    if (i < n) out[i] = f2bf(in[i]);
}

__global__ __launch_bounds__(256)
void transpose_bf16(const float* __restrict__ W, unsigned short* __restrict__ WT,
                    int K, int N)
{
    __shared__ unsigned short t[32][33];
    int n0 = blockIdx.x * 32, k0 = blockIdx.y * 32;
    int tx = threadIdx.x & 31, ty = threadIdx.x >> 5;     // 32 x 8
#pragma unroll
    for (int i = 0; i < 32; i += 8)
        t[ty + i][tx] = f2bf(W[(size_t)(k0 + ty + i) * N + n0 + tx]);
    __syncthreads();
#pragma unroll
    for (int i = 0; i < 32; i += 8)
        WT[(size_t)(n0 + ty + i) * K + k0 + tx] = t[tx][ty + i];
}

// ---------------------------------------------------------------------------
// bf16-WMMA GEMM with TDM double-buffered staging.
//   C[M,N] = A[M,K] * BT[N,K]^T ; A,BT bf16, C fp32 or bf16.
// Block: 256 thr (8 waves). Tile 128(M) x 64(N), K-slice 32.
// Wave0 drives the TDM; tile i+1 streams while all waves compute tile i.
// ---------------------------------------------------------------------------
template <bool OUT_F32>
__global__ __launch_bounds__(256)
void gemm_bf16_tdm(const unsigned short* __restrict__ A,
                   const unsigned short* __restrict__ BT,
                   void* __restrict__ Cptr, int M, int N, int K)
{
    constexpr int BM = 128, BN = 64, BK = 32;
    __shared__ __align__(16) unsigned short lA[2][BM * BK]; // [m][k] 2x8KB
    __shared__ __align__(16) unsigned short lB[2][BN * BK]; // [n][k] 2x4KB

    const int tid  = threadIdx.x;
    const int wave = tid >> 5;
    const int lane = tid & 31;
    const int hl   = lane & 15;
    const bool lo  = lane < 16;
    const int tileN = blockIdx.x * BN;
    const int tileM = blockIdx.y * BM;

    const unsigned short* Ag = A  + (size_t)tileM * K;
    const unsigned short* Bg = BT + (size_t)tileN * K;

    if (wave == 0) {   // prologue: stream tile 0 into buffer 0
        tdm_load_2d(lds_off(&lA[0][0]), Ag, BK, BM, (unsigned long long)K);
        tdm_load_2d(lds_off(&lB[0][0]), Bg, BK, BN, (unsigned long long)K);
    }

    v8f acc[4];
#pragma unroll
    for (int t = 0; t < 4; ++t) acc[t] = v8f{};

    int buf = 0;
    for (int k0 = 0; k0 < K; k0 += BK, buf ^= 1) {
        if (wave == 0) {
            if (k0 + BK < K) {     // prefetch tile i+1 into the other buffer
                tdm_load_2d(lds_off(&lA[buf ^ 1][0]), Ag + (k0 + BK), BK, BM,
                            (unsigned long long)K);
                tdm_load_2d(lds_off(&lB[buf ^ 1][0]), Bg + (k0 + BK), BK, BN,
                            (unsigned long long)K);
                __builtin_amdgcn_s_wait_tensorcnt(2);  // tile i done (in-order)
            } else {
                __builtin_amdgcn_s_wait_tensorcnt(0);
            }
        }
        __syncthreads();

        // A fragment (16x32 bf16): lanes0-15 K{0..7,16..23}, lanes16-31 K{8..15,24..31}
        {
            int m  = 16 * wave + hl;
            int ka = lo ? 0 : 8;
            const unsigned short* a = &lA[buf][m * BK];
            v8u alo = *(const v8u*)&a[ka];
            v8u ahi = *(const v8u*)&a[ka + 16];
            v16bf aF = as_bf(cat8(alo, ahi));
#pragma unroll
            for (int t = 0; t < 4; ++t) {
                int n  = 16 * t + hl;
                int kb = lo ? 0 : 8;           // B: lane=col, K {0..7,16..23} / {8..15,24..31}
                const unsigned short* bp = &lB[buf][n * BK];
                v8u blo = *(const v8u*)&bp[kb];
                v8u bhi = *(const v8u*)&bp[kb + 16];
                v16bf bF = as_bf(cat8(blo, bhi));
                acc[t] = __builtin_amdgcn_wmma_f32_16x16x32_bf16(
                    false, aF, false, bF, (short)0, acc[t], false, false);
            }
        }
        __syncthreads();   // readers done before TDM overwrites this buffer
    }

    // epilogue: C layout = VGPR r -> row r (lanes<16) / r+8 (lanes>=16), col = lane%16
#pragma unroll
    for (int t = 0; t < 4; ++t) {
        int col = tileN + 16 * t + hl;
#pragma unroll
        for (int r = 0; r < 8; ++r) {
            int row = tileM + 16 * wave + r + (lo ? 0 : 8);
            float v = acc[t][r];
            if constexpr (OUT_F32)
                ((float*)Cptr)[(size_t)row * N + col] = v;
            else
                ((unsigned short*)Cptr)[(size_t)row * N + col] = f2bf(v);
        }
    }
}

// ---------------------------------------------------------------------------
// RoPE on q,k halves of the bf16 qkv buffer; each thread owns a (d, d+64)
// rotation pair -> race-free in-place update.
// ---------------------------------------------------------------------------
__global__ __launch_bounds__(256)
void rope_kernel(unsigned short* __restrict__ qkv,
                 const float* __restrict__ cosT, const float* __restrict__ sinT)
{
    int gid = blockIdx.x * blockDim.x + threadIdx.x;   // ROWS*2048 threads
    int row   = gid >> 11;
    int rem   = gid & 2047;
    int which = rem >> 10;        // 0 = q, 1 = k
    int p     = rem & 1023;
    int h = p >> 6, d = p & 63;
    int s = row & (SEQ - 1);
    size_t base = (size_t)row * QKV_N + which * D_MODEL + h * HEAD_DIM + d;
    float u1 = bf2f(qkv[base]);
    float u2 = bf2f(qkv[base + 64]);
    float c1 = cosT[s * HEAD_DIM + d],      s1 = sinT[s * HEAD_DIM + d];
    float c2 = cosT[s * HEAD_DIM + d + 64], s2 = sinT[s * HEAD_DIM + d + 64];
    qkv[base]      = f2bf(u1 * c1 - u2 * s1);
    qkv[base + 64] = f2bf(u2 * c2 + u1 * s2);
}

// ---------------------------------------------------------------------------
// Causal flash attention per (b, h, 64-row q tile). 128 thr = 4 waves.
// Q (once) and each K tile stream in via TDM; V is staged transposed by hand
// (overlaps the K DMA). QK^T: 4 chained bf16 WMMAs; online softmax with
// shfl_xor row reductions; P re-laid out C->A via per-wave LDS; P*V: 8 WMMAs.
// ---------------------------------------------------------------------------
__global__ __launch_bounds__(128)
void flash_attn(const unsigned short* __restrict__ qkv,
                unsigned short* __restrict__ ctx)
{
    constexpr int QT = 64;
    const int tid  = threadIdx.x;
    const int wave = tid >> 5;
    const int lane = tid & 31;
    const int hl   = lane & 15;
    const bool lo  = lane < 16;
    const int qb = blockIdx.x * QT;
    const int h  = blockIdx.y;
    const int b  = blockIdx.z;

    __shared__ __align__(16) unsigned short lQ[QT * HEAD_DIM];   // [m][d]  16KB
    __shared__ __align__(16) unsigned short lK[32 * HEAD_DIM];   // [kv][d]  8KB
    __shared__ __align__(16) unsigned short lV[HEAD_DIM * 32];   // [d][kv]T 8KB
    __shared__ __align__(16) unsigned short lP[4][16 * 32];      // per-wave 4KB

    const size_t rs = QKV_N;
    const unsigned short* Qg = qkv + (size_t)(b * SEQ) * rs + h * HEAD_DIM;
    const unsigned short* Kg = Qg + D_MODEL;
    const unsigned short* Vg = Qg + 2 * D_MODEL;

    if (wave == 0)   // Q tile: 64 rows x 128, row pitch 6144 -> packed LDS
        tdm_load_2d(lds_off(lQ), Qg + (size_t)qb * rs, HEAD_DIM, QT,
                    (unsigned long long)rs);

    float mrow[8], srow[8];
#pragma unroll
    for (int r = 0; r < 8; ++r) { mrow[r] = -3.0e38f; srow[r] = 0.f; }
    v8f acc[8];
#pragma unroll
    for (int t = 0; t < 8; ++t) acc[t] = v8f{};

    const float scale = 0.08838834764831845f;   // 1/sqrt(128)
    const int kvEnd = qb + QT;                  // causal bound for this block

    for (int kv0 = 0; kv0 < kvEnd; kv0 += 32) {
        __syncthreads();   // previous-iter readers of lK/lV done
        if (wave == 0)     // K tile via TDM (direct row copy)
            tdm_load_2d(lds_off(lK), Kg + (size_t)kv0 * rs, HEAD_DIM, 32,
                        (unsigned long long)rs);
        for (int i = 0; i < 32; ++i) {          // V staged transposed [d][kv]
            int lin = i * 128 + tid;
            int kv = lin >> 7, d = lin & 127;
            lV[d * 32 + kv] = Vg[(size_t)(kv0 + kv) * rs + d];
        }
        if (wave == 0)
            __builtin_amdgcn_s_wait_tensorcnt(0);   // covers Q on first iter too
        __syncthreads();

        // ---- scores: two 16-col subtiles, contraction over head_dim=128
        v8f sc[2];
#pragma unroll
        for (int t = 0; t < 2; ++t) {
            sc[t] = v8f{};
#pragma unroll
            for (int kc = 0; kc < 4; ++kc) {
                int m   = 16 * wave + hl;
                int off = kc * 32 + (lo ? 0 : 8);
                v8u qlo = *(const v8u*)&lQ[m * HEAD_DIM + off];
                v8u qhi = *(const v8u*)&lQ[m * HEAD_DIM + off + 16];
                v16bf qF = as_bf(cat8(qlo, qhi));
                int n  = 16 * t + hl;
                int kb = kc * 32 + (lo ? 0 : 8);
                v8u klo = *(const v8u*)&lK[n * HEAD_DIM + kb];
                v8u khi = *(const v8u*)&lK[n * HEAD_DIM + kb + 16];
                v16bf kF = as_bf(cat8(klo, khi));
                sc[t] = __builtin_amdgcn_wmma_f32_16x16x32_bf16(
                    false, qF, false, kF, (short)0, sc[t], false, false);
            }
        }

        // ---- scale + causal mask + row max over both subtiles
        float tmax[8];
#pragma unroll
        for (int r = 0; r < 8; ++r) tmax[r] = -3.0e38f;
#pragma unroll
        for (int t = 0; t < 2; ++t) {
            int col = kv0 + 16 * t + hl;
#pragma unroll
            for (int r = 0; r < 8; ++r) {
                int row = qb + 16 * wave + r + (lo ? 0 : 8);
                float v = sc[t][r] * scale;
                v = (col <= row) ? v : -3.0e38f;
                sc[t][r] = v;
                tmax[r] = fmaxf(tmax[r], v);
            }
        }
#pragma unroll
        for (int r = 0; r < 8; ++r) {           // reduce a row (16 lanes)
            float v = tmax[r];
#pragma unroll
            for (int msk = 1; msk < 16; msk <<= 1)
                v = fmaxf(v, __shfl_xor(v, msk, 32));
            tmax[r] = v;
        }

        float corr[8];
#pragma unroll
        for (int r = 0; r < 8; ++r) {
            float nm = fmaxf(mrow[r], tmax[r]);
            corr[r] = __expf(mrow[r] - nm);
            mrow[r] = nm;
        }

        float rsum[8];
#pragma unroll
        for (int r = 0; r < 8; ++r) rsum[r] = 0.f;
#pragma unroll
        for (int t = 0; t < 2; ++t) {
#pragma unroll
            for (int r = 0; r < 8; ++r) {
                float v = sc[t][r];
                float p = (v <= -1.0e37f) ? 0.f : __expf(v - mrow[r]);
                rsum[r] += p;
                int rl = r + (lo ? 0 : 8);
                lP[wave][rl * 32 + 16 * t + hl] = f2bf(p);
            }
        }
#pragma unroll
        for (int r = 0; r < 8; ++r) {
            float v = rsum[r];
#pragma unroll
            for (int msk = 1; msk < 16; msk <<= 1)
                v += __shfl_xor(v, msk, 32);
            srow[r] = srow[r] * corr[r] + v;
        }
#pragma unroll
        for (int t = 0; t < 8; ++t)
#pragma unroll
            for (int r = 0; r < 8; ++r)
                acc[t][r] *= corr[r];

        asm volatile("s_wait_dscnt 0" ::: "memory");  // lP cross-lane handoff

        // ---- P (A-frag, 16x32) x V (B-frag, 32x16 per head-dim subtile)
        {
            int off = lo ? 0 : 8;
            v8u plo = *(const v8u*)&lP[wave][hl * 32 + off];
            v8u phi = *(const v8u*)&lP[wave][hl * 32 + off + 16];
            v16bf pF = as_bf(cat8(plo, phi));
#pragma unroll
            for (int t = 0; t < 8; ++t) {
                int n  = 16 * t + hl;
                int kb = lo ? 0 : 8;
                v8u vlo = *(const v8u*)&lV[n * 32 + kb];
                v8u vhi = *(const v8u*)&lV[n * 32 + kb + 16];
                v16bf vF = as_bf(cat8(vlo, vhi));
                acc[t] = __builtin_amdgcn_wmma_f32_16x16x32_bf16(
                    false, pF, false, vF, (short)0, acc[t], false, false);
            }
        }
    }

    // ---- epilogue: ctx = acc / rowsum, bf16 into workspace [ROWS][D_MODEL]
#pragma unroll
    for (int t = 0; t < 8; ++t) {
        int col = h * HEAD_DIM + 16 * t + hl;
#pragma unroll
        for (int r = 0; r < 8; ++r) {
            int row = b * SEQ + qb + 16 * wave + r + (lo ? 0 : 8);
            ctx[(size_t)row * D_MODEL + col] = f2bf(acc[t][r] / srow[r]);
        }
    }
}

// ---------------------------------------------------------------------------
extern "C" void kernel_launch(void* const* d_in, const int* in_sizes, int n_in,
                              void* d_out, int out_size, void* d_ws, size_t ws_size,
                              hipStream_t stream)
{
    const float* x    = (const float*)d_in[0];
    const float* cosT = (const float*)d_in[1];
    const float* sinT = (const float*)d_in[2];
    const float* wqkv = (const float*)d_in[3];
    const float* wo   = (const float*)d_in[4];
    float* out = (float*)d_out;

    unsigned short* xb    = (unsigned short*)d_ws;                 // ROWS*D_MODEL
    unsigned short* wqkvT = xb    + (size_t)ROWS * D_MODEL;        // QKV_N*D_MODEL
    unsigned short* woT   = wqkvT + (size_t)QKV_N * D_MODEL;       // D_MODEL^2
    unsigned short* qkv   = woT   + (size_t)D_MODEL * D_MODEL;     // ROWS*QKV_N
    unsigned short* ctx   = qkv   + (size_t)ROWS * QKV_N;          // ROWS*D_MODEL

    // 0) one-time bf16 conversion (+ weight transpose to [N][K])
    cvt_bf16<<<(ROWS * D_MODEL) / 256, 256, 0, stream>>>(x, xb, ROWS * D_MODEL);
    transpose_bf16<<<dim3(QKV_N / 32, D_MODEL / 32), 256, 0, stream>>>(
        wqkv, wqkvT, D_MODEL, QKV_N);
    transpose_bf16<<<dim3(D_MODEL / 32, D_MODEL / 32), 256, 0, stream>>>(
        wo, woT, D_MODEL, D_MODEL);

    // 1) QKV projection (TDM-staged bf16 WMMA) -> bf16 qkv
    dim3 g1(QKV_N / 64, ROWS / 128);
    gemm_bf16_tdm<false><<<g1, 256, 0, stream>>>(xb, wqkvT, qkv,
                                                 ROWS, QKV_N, D_MODEL);
    // 2) RoPE on q,k
    rope_kernel<<<(ROWS * 2048) / 256, 256, 0, stream>>>(qkv, cosT, sinT);

    // 3) causal flash attention
    dim3 g2(SEQ / 64, N_HEADS, BATCH);
    flash_attn<<<g2, 128, 0, stream>>>(qkv, ctx);

    // 4) output projection -> fp32 out
    dim3 g3(D_MODEL / 64, ROWS / 128);
    gemm_bf16_tdm<true><<<g3, 256, 0, stream>>>(ctx, woT, out,
                                                ROWS, D_MODEL, D_MODEL);
}
